// HyperbolicKuramotoAttention_25615184953681
// MI455X (gfx1250) — compile-verified, matrix operand order checked by reading, not run on hardware
//
#include <hip/hip_runtime.h>
#include <math.h>

typedef __attribute__((ext_vector_type(16))) __bf16 v16bf;
typedef __attribute__((ext_vector_type(8)))  __bf16 v8bf;
typedef __attribute__((ext_vector_type(8)))  float  v8f;

#define D_MODEL 1024
#define N_HEADS 16
#define HD      64
#define SEQ     1024
#define BATCH   2
#define BL      (BATCH*SEQ)

// ---------------------------------------------------------------------------
// D = A(16x32 bf16) * B(32x16 bf16) + C(16x16 f32), wave32 WMMA
__device__ __forceinline__ v8f wmma_bf16(v16bf a, v16bf b, v8f c) {
  return __builtin_amdgcn_wmma_f32_16x16x32_bf16(
      /*neg_a=*/false, a, /*neg_b=*/false, b,
      /*c_mod=*/(short)0, c, /*reuse_a=*/false, /*reuse_b=*/false);
}

// Load one 16-bit A/B-style fragment from a row-major bf16 matrix.
// Lane `sel` (= lane&15) supplies row `sel`; group g (= lane>>4) selects the
// K half. Per the ISA layout, the 16 per-lane elements are two contiguous
// 8-element runs at k0+8g and k0+16+8g -> two 16-byte loads.
__device__ __forceinline__ v16bf load_frag(const __bf16* __restrict__ base,
                                           int row_stride, int k0, int sel, int g) {
  const __bf16* p = base + (size_t)sel * row_stride + k0 + 8 * g;
  v8bf lo = *(const v8bf*)(p);
  v8bf hi = *(const v8bf*)(p + 16);
  v16bf f;
#pragma unroll
  for (int i = 0; i < 8; ++i) { f[i] = lo[i]; f[8 + i] = hi[i]; }
  return f;
}

// ---------------------------------------------------------------------------
__global__ void cvt_f32_bf16_kernel(const float* __restrict__ src,
                                    __bf16* __restrict__ dst, int n) {
  int i = blockIdx.x * blockDim.x + threadIdx.x;
  if (i < n) dst[i] = (__bf16)src[i];
}

// ---------------------------------------------------------------------------
// QKV projection: Y = X @ W^T + b. One wave per 16x16 tile; blockIdx.y picks
// which of Q/K/V.  Q,K stored bf16 [B,H,L,hd]; V stored transposed [B,H,hd,S].
__global__ __launch_bounds__(256) void qkv_kernel(
    const __bf16* __restrict__ Xb,
    const __bf16* __restrict__ Wqb, const __bf16* __restrict__ Wkb,
    const __bf16* __restrict__ Wvb,
    const float* __restrict__ bq, const float* __restrict__ bk,
    const float* __restrict__ bv,
    __bf16* __restrict__ Qb, __bf16* __restrict__ Kb, __bf16* __restrict__ Vt) {
  int lane = threadIdx.x & 31;
  int wv   = threadIdx.x >> 5;
  int which = blockIdx.y;
  const __bf16* W   = (which == 0) ? Wqb : (which == 1) ? Wkb : Wvb;
  const float*  bia = (which == 0) ? bq  : (which == 1) ? bk  : bv;

  int tile = blockIdx.x * 8 + wv;
  int nt = tile & 63, mt = tile >> 6;
  int m0 = mt * 16, n0 = nt * 16;
  int sel = lane & 15, g = lane >> 4;

  const __bf16* Abase = Xb + (size_t)m0 * D_MODEL;
  const __bf16* Bbase = W  + (size_t)n0 * D_MODEL;

  v8f c = {};
#pragma unroll 4
  for (int k0 = 0; k0 < D_MODEL; k0 += 32) {
    v16bf a = load_frag(Abase, D_MODEL, k0, sel, g);
    v16bf b = load_frag(Bbase, D_MODEL, k0, sel, g);
    c = wmma_bf16(a, b, c);
  }

  int feat = n0 + sel;
  int h = feat >> 6, d = feat & 63;
  float bs = bia[feat];
#pragma unroll
  for (int r = 0; r < 8; ++r) {
    int row = m0 + r + 8 * g;          // global row in [0, B*L)
    int b_  = row >> 10;
    int l   = row & (SEQ - 1);
    float v = c[r] + bs;
    if (which == 2)
      Vt[(((size_t)b_ * N_HEADS + h) * HD + d) * SEQ + l] = (__bf16)v;
    else if (which == 1)
      Kb[(((size_t)b_ * N_HEADS + h) * SEQ + l) * HD + d] = (__bf16)v;
    else
      Qb[(((size_t)b_ * N_HEADS + h) * SEQ + l) * HD + d] = (__bf16)v;
  }
}

// ---------------------------------------------------------------------------
__global__ void norms_kernel(const __bf16* __restrict__ Qb,
                             const __bf16* __restrict__ Kb,
                             float* __restrict__ qn, float* __restrict__ kn) {
  int i = blockIdx.x * blockDim.x + threadIdx.x;   // (b,h,l) flat, 32768 total
  const __bf16* q = Qb + (size_t)i * HD;
  const __bf16* k = Kb + (size_t)i * HD;
  float sq = 0.f, sk = 0.f;
#pragma unroll 8
  for (int d = 0; d < HD; ++d) {
    float x = (float)q[d]; sq += x * x;
    float y = (float)k[d]; sk += y * y;
  }
  qn[i] = sq;
  kn[i] = sk;
}

// ---------------------------------------------------------------------------
// One Euler step of Kuramoto; writes phases [B,H] and order [B] directly to
// the output tail, phase_mod [B,H] to workspace.  32 threads.
__global__ void kuramoto_kernel(const float* __restrict__ omega,
                                const float* __restrict__ theta0,
                                const float* __restrict__ cK,
                                float* __restrict__ phases_out,
                                float* __restrict__ order_out,
                                float* __restrict__ pmod) {
  int t = threadIdx.x;            // 0..31
  int b = t >> 4, i = t & 15;
  float th = theta0[i];
  float s = 0.f;
  for (int j = 0; j < N_HEADS; ++j) s += cK[i * N_HEADS + j] * sinf(theta0[j] - th);
  float ph = th + 0.1f * (omega[i] + s * (1.0f / N_HEADS));
  __shared__ float sph[32];
  sph[t] = ph;
  __syncthreads();
  float pm = 0.f;
  for (int j = 0; j < N_HEADS; ++j) pm += cosf(ph - sph[b * 16 + j]);
  pmod[t] = pm * (1.0f / N_HEADS);
  phases_out[t] = ph;
  if (i == 0) {
    float cs = 0.f, ss = 0.f;
    for (int j = 0; j < N_HEADS; ++j) { cs += cosf(sph[b * 16 + j]); ss += sinf(sph[b * 16 + j]); }
    cs *= (1.0f / N_HEADS); ss *= (1.0f / N_HEADS);
    order_out[b] = sqrtf(cs * cs + ss * ss);
  }
}

// ---------------------------------------------------------------------------
// Flash-style hyperbolic attention.  One wave = 16 queries of one (b,h).
// Keys streamed in chunks of 32; QK^T via WMMA, score math in f32 on the
// C-layout tile, probabilities staged through LDS (C-layout -> A-layout),
// P*V via WMMA.
//
// Key simplifications vs the literal reference:
//   exp(-log(arg+sqrt(arg^2-1))) == 1/(arg+sqrt(arg^2-1))  -> no log/exp
//   pre-softmax scores are in [-1,1] (|pm|<=1, dist>=0), so exp() cannot
//   overflow and the softmax max-subtraction is unnecessary -> no running
//   max, no O rescaling, single row-sum reduction at the end.
__global__ __launch_bounds__(256) void attn_kernel(
    const __bf16* __restrict__ Qb, const __bf16* __restrict__ Kb,
    const __bf16* __restrict__ Vt, const float* __restrict__ qn,
    const float* __restrict__ kn, const float* __restrict__ pmod,
    const int* __restrict__ mask, __bf16* __restrict__ ctxb) {
  __shared__ __align__(16) __bf16 Plds[8][16][32];   // per-wave 16x32 P tile

  int lane = threadIdx.x & 31;
  int wv   = threadIdx.x >> 5;
  int id = blockIdx.x * 8 + wv;      // 2048 waves total
  int mt = id & 63;
  int h  = (id >> 6) & 15;
  int b  = id >> 10;
  int l0 = mt * 16;
  int sel = lane & 15, g = lane >> 4;

  size_t bh = (size_t)b * N_HEADS + h;
  const __bf16* Qh = Qb + bh * SEQ * HD;
  const __bf16* Kh = Kb + bh * SEQ * HD;
  const __bf16* Vh = Vt + bh * HD * SEQ;
  const float* qnh = qn + bh * SEQ;
  const float* knh = kn + bh * SEQ;
  const int*   mb  = mask + (size_t)b * SEQ * SEQ + (size_t)l0 * SEQ;
  float pm = pmod[bh];

  // Q fragments: 16x64 = two 16x32 k-slices, loaded once.
  v16bf qa0 = load_frag(Qh + (size_t)l0 * HD, HD, 0,  sel, g);
  v16bf qa1 = load_frag(Qh + (size_t)l0 * HD, HD, 32, sel, g);

  float qrow[8], qc1[8];               // ||q||^2 and (1 - clamp(||q||^2))
#pragma unroll
  for (int r = 0; r < 8; ++r) {
    qrow[r] = qnh[l0 + r + 8 * g];
    qc1[r]  = 1.0f - fminf(qrow[r], 0.99f);
  }

  float psum[8];                       // per-lane partial row sums of exp(s)
#pragma unroll
  for (int r = 0; r < 8; ++r) psum[r] = 0.f;
  v8f O[4] = {};

  for (int s0 = 0; s0 < SEQ; s0 += 32) {
    if (s0 + 32 < SEQ)
      __builtin_prefetch(Kh + (size_t)(s0 + 32) * HD + sel * 4, 0, 0);

#pragma unroll
    for (int t = 0; t < 2; ++t) {
      int sbase = s0 + 16 * t;
      v16bf kb0 = load_frag(Kh + (size_t)sbase * HD, HD, 0,  sel, g);
      v16bf kb1 = load_frag(Kh + (size_t)sbase * HD, HD, 32, sel, g);
      v8f dot = {};
      dot = wmma_bf16(qa0, kb0, dot);
      dot = wmma_bf16(qa1, kb1, dot);

      float kv  = knh[sbase + sel];
      float kc1 = 1.0f - fminf(kv, 0.99f);
#pragma unroll
      for (int r = 0; r < 8; ++r) {
        float diff  = fmaxf(qrow[r] + kv - 2.0f * dot[r], 0.0f);
        float denom = qc1[r] * kc1 + 1e-8f;
        float arg   = 1.0f + 2.0f * diff / denom;
        // exp(-acosh(arg)) == 1/(arg + sqrt(arg^2-1))
        float w = 1.0f / (arg + sqrtf(fmaxf(arg * arg - 1.0f, 1e-8f)));
        float s = pm * w;              // s in [-1, 1] -> exp is safe
        float p = expf(s);
        if (mb[(size_t)(r + 8 * g) * SEQ + sbase + sel] == 0) p = 0.0f;
        psum[r] += p;
        Plds[wv][r + 8 * g][16 * t + sel] = (__bf16)p;
      }
    }

    // wave-synchronous LDS handoff (same-wave DS ops are in order; fence the
    // compiler and drain DScnt before the A-layout reads)
    asm volatile("s_wait_dscnt 0" ::: "memory");

    const v8bf* prow = (const v8bf*)(&Plds[wv][sel][0]);
    v8bf plo = prow[g];        // k = 8g .. 8g+7
    v8bf phi = prow[2 + g];    // k = 16+8g .. 16+8g+7
    v16bf pf;
#pragma unroll
    for (int i = 0; i < 8; ++i) { pf[i] = plo[i]; pf[8 + i] = phi[i]; }

#pragma unroll
    for (int dt = 0; dt < 4; ++dt) {
      v16bf vf = load_frag(Vh + (size_t)(dt * 16) * SEQ, SEQ, s0, sel, g);
      O[dt] = wmma_bf16(pf, vf, O[dt]);
    }
  }

  // single row-sum reduction across the 16 lanes of each half-wave
  float inv[8];
#pragma unroll
  for (int r = 0; r < 8; ++r) {
    float v = psum[r];
    v += __shfl_xor(v, 1); v += __shfl_xor(v, 2);
    v += __shfl_xor(v, 4); v += __shfl_xor(v, 8);
    inv[r] = 1.0f / v;
  }

  // normalize and store ctx in [B, L, H*hd] (bf16) for the output GEMM
  __bf16* cb = ctxb + (size_t)b * SEQ * D_MODEL + (size_t)h * HD;
#pragma unroll
  for (int dt = 0; dt < 4; ++dt)
#pragma unroll
    for (int r = 0; r < 8; ++r)
      cb[(size_t)(l0 + r + 8 * g) * D_MODEL + dt * 16 + sel] = (__bf16)(O[dt][r] * inv[r]);
}

// ---------------------------------------------------------------------------
__global__ __launch_bounds__(256) void outproj_kernel(
    const __bf16* __restrict__ ctxb, const __bf16* __restrict__ Wob,
    const float* __restrict__ bo, float* __restrict__ out) {
  int lane = threadIdx.x & 31;
  int wv   = threadIdx.x >> 5;
  int tile = blockIdx.x * 8 + wv;
  int nt = tile & 63, mt = tile >> 6;
  int m0 = mt * 16, n0 = nt * 16;
  int sel = lane & 15, g = lane >> 4;

  const __bf16* Ab = ctxb + (size_t)m0 * D_MODEL;
  const __bf16* Bb = Wob  + (size_t)n0 * D_MODEL;
  v8f c = {};
#pragma unroll 4
  for (int k0 = 0; k0 < D_MODEL; k0 += 32)
    c = wmma_bf16(load_frag(Ab, D_MODEL, k0, sel, g),
                  load_frag(Bb, D_MODEL, k0, sel, g), c);

  float bs = bo[n0 + sel];
#pragma unroll
  for (int r = 0; r < 8; ++r)
    out[(size_t)(m0 + r + 8 * g) * D_MODEL + n0 + sel] = c[r] + bs;
}

// ---------------------------------------------------------------------------
extern "C" void kernel_launch(void* const* d_in, const int* in_sizes, int n_in,
                              void* d_out, int out_size, void* d_ws, size_t ws_size,
                              hipStream_t stream) {
  (void)in_sizes; (void)n_in; (void)out_size; (void)ws_size;

  const float* X      = (const float*)d_in[0];
  const int*   mask   = (const int*)d_in[1];
  const float* Wq     = (const float*)d_in[2];
  const float* bq     = (const float*)d_in[3];
  const float* Wk     = (const float*)d_in[4];
  const float* bk     = (const float*)d_in[5];
  const float* Wv     = (const float*)d_in[6];
  const float* bv     = (const float*)d_in[7];
  const float* Wo     = (const float*)d_in[8];
  const float* bo     = (const float*)d_in[9];
  const float* omega  = (const float*)d_in[10];
  const float* theta0 = (const float*)d_in[11];
  const float* cK     = (const float*)d_in[12];
  float* out = (float*)d_out;

  // workspace carve-up (bf16 unless noted)
  __bf16* Xb   = (__bf16*)d_ws;
  __bf16* Wqb  = Xb   + (size_t)BL * D_MODEL;
  __bf16* Wkb  = Wqb  + (size_t)D_MODEL * D_MODEL;
  __bf16* Wvb  = Wkb  + (size_t)D_MODEL * D_MODEL;
  __bf16* Wob  = Wvb  + (size_t)D_MODEL * D_MODEL;
  __bf16* Qb   = Wob  + (size_t)D_MODEL * D_MODEL;
  __bf16* Kb   = Qb   + (size_t)BL * D_MODEL;
  __bf16* Vt   = Kb   + (size_t)BL * D_MODEL;
  __bf16* ctxb = Vt   + (size_t)BL * D_MODEL;
  float*  qn   = (float*)(ctxb + (size_t)BL * D_MODEL);
  float*  kn   = qn + BATCH * N_HEADS * SEQ;
  float*  pmod = kn + BATCH * N_HEADS * SEQ;

  const int NXT = BL * D_MODEL;           // 2M elements
  const int NWT = D_MODEL * D_MODEL;      // 1M elements
  cvt_f32_bf16_kernel<<<NXT / 256, 256, 0, stream>>>(X,  Xb,  NXT);
  cvt_f32_bf16_kernel<<<NWT / 256, 256, 0, stream>>>(Wq, Wqb, NWT);
  cvt_f32_bf16_kernel<<<NWT / 256, 256, 0, stream>>>(Wk, Wkb, NWT);
  cvt_f32_bf16_kernel<<<NWT / 256, 256, 0, stream>>>(Wv, Wvb, NWT);
  cvt_f32_bf16_kernel<<<NWT / 256, 256, 0, stream>>>(Wo, Wob, NWT);

  qkv_kernel<<<dim3(1024, 3), 256, 0, stream>>>(Xb, Wqb, Wkb, Wvb, bq, bk, bv,
                                                Qb, Kb, Vt);
  norms_kernel<<<128, 256, 0, stream>>>(Qb, Kb, qn, kn);
  kuramoto_kernel<<<1, 32, 0, stream>>>(omega, theta0, cK,
                                        out + (size_t)BL * D_MODEL,
                                        out + (size_t)BL * D_MODEL + 32, pmod);
  attn_kernel<<<256, 256, 0, stream>>>(Qb, Kb, Vt, qn, kn, pmod, mask, ctxb);
  outproj_kernel<<<1024, 256, 0, stream>>>(ctxb, Wob, bo, out);
}
